// AttentionSACModel_34179349741800
// MI455X (gfx1250) — matrix operand order repaired and checked by reading, not compiled, not run on hardware
//
#include <hip/hip_runtime.h>

typedef __attribute__((ext_vector_type(16))) _Float16 v16h;
typedef __attribute__((ext_vector_type(8)))  _Float16 v8h;
typedef __attribute__((ext_vector_type(8)))  float    v8f;

#define OWN_DIM    7
#define INT_DIM    7
#define N_INTR     20
#define NH         3
#define DD         42
#define HSTRIDE    48    /* head-padded column stride (DD padded to 48) */
#define TOT        126
#define TOTP       160   /* padded K for w_ownp / w_attn (3*48 -> 160 for 5 tiles) */
#define ATTN_DIM   128
#define HID        256
#define NOUT       4
#define BATCH      32768
#define OBS_STRIDE 147   /* 7 + 20*7 */

#define ROWS_PER_WAVE   16
#define WAVES_PER_BLOCK 4
#define ROWS_PER_BLOCK  64
#define SCR_LD          160  /* head-padded embedding buffer, 16B-aligned rows */
#define XB_LD           296  /* 0:127 x-lower, 128:287 ctx (head-padded) / attn_vec */
#define CTX_BASE        128

/* workspace offsets in halves; each B-tile is 512 halves (32Kx16N) */
#define OFF_WOWN   0u        /* 1x8  tiles =  4096 */
#define OFF_WINT   4096u     /* 1x8  tiles =  4096 */
#define OFF_WQ     8192u     /* 3 heads x (2x3 tiles = 3072) = 9216 */
#define OFF_WK     17408u
#define OFF_WV     26624u
#define OFF_WOWNP  35840u    /* 5x8  tiles = 20480 (head-padded K'=160) */
#define OFF_WATTN  56320u    /* 5x8  tiles = 20480 (head-padded K'=160) */
#define OFF_WH1    76800u    /* 8x16 tiles = 65536 */
#define OFF_WH2    142336u   /* 8x16 tiles = 65536 */
#define OFF_WOUT   207872u   /* 8x1  tiles =  4096 */

__device__ __forceinline__ v8f zero8() {
    v8f z;
#pragma unroll
    for (int i = 0; i < 8; ++i) z[i] = 0.0f;
    return z;
}
__device__ __forceinline__ v8h zero8h() {
    v8h z;
#pragma unroll
    for (int i = 0; i < 8; ++i) z[i] = (_Float16)0.0f;
    return z;
}
__device__ __forceinline__ v16h zero16() {
    v16h z;
#pragma unroll
    for (int i = 0; i < 16; ++i) z[i] = (_Float16)0.0f;
    return z;
}

/* cheap tanh: single-trans v_tanh_f32 if the builtin exists, else exp+rcp */
__device__ __forceinline__ float fast_tanh(float x) {
#if defined(__gfx1250__) && __has_builtin(__builtin_amdgcn_tanhf)
    return __builtin_amdgcn_tanhf(x);
#else
    const float e = __expf(2.0f * x);
    return 1.0f - 2.0f * __builtin_amdgcn_rcpf(e + 1.0f);
#endif
}

/* WMMA: D = A(16x32 f16) * B(32x16 f16) + C(16x16 f32) */
__device__ __forceinline__ v8f wmma_f16(v16h a, v16h b, v8f c) {
    return __builtin_amdgcn_wmma_f32_16x16x32_f16(false, a, false, b, (short)0, c, false, false);
}

/* B fragment: tiles are pre-packed so each lane reads 16 contiguous halves */
__device__ __forceinline__ v16h ldb(const _Float16* __restrict__ wt, int tile, int lane) {
    return *reinterpret_cast<const v16h*>(wt + (size_t)tile * 512 + lane * 16);
}

/* A fragment from LDS: full 32-wide 16B-aligned slice -> 2x ds_load_b128.
   All call sites now satisfy alignment; pad columns are kept zero in LDS. */
__device__ __forceinline__ v16h lda(const _Float16* buf, int ld, int lane, int colbase) {
    const int row = lane & 15;
    const int k0  = (lane >> 4) * 16;
    const _Float16* p = buf + row * ld + colbase;
    const v8h lo = *reinterpret_cast<const v8h*>(p + k0);
    const v8h hi = *reinterpret_cast<const v8h*>(p + k0 + 8);
    return __builtin_shufflevector(lo, hi, 0, 1, 2, 3, 4, 5, 6, 7,
                                   8, 9, 10, 11, 12, 13, 14, 15);
}

/* C accumulator -> LDS (f16). row = i + 8*(lane>>4), col = lane&15 */
__device__ __forceinline__ void sta(_Float16* buf, int ld, int lane, int colbase,
                                    int nvalid, v8f acc) {
    const int col = lane & 15;
    const int rb  = (lane >> 4) * 8;
    if (col < nvalid) {
#pragma unroll
        for (int i = 0; i < 8; ++i)
            buf[(rb + i) * ld + colbase + col] = (_Float16)acc[i];
    }
}

/* C accumulator -> LDS with precomputed per-lane column map (-1 = skip) */
__device__ __forceinline__ void sta_map(_Float16* buf, int ld, int lane, int mcol, v8f acc) {
    const int rb = (lane >> 4) * 8;
    if (mcol >= 0) {
#pragma unroll
        for (int i = 0; i < 8; ++i)
            buf[(rb + i) * ld + mcol] = (_Float16)acc[i];
    }
}

/* -------- prep: repack f32 [K][N] weights into WMMA B-fragment tiles -------- */
__global__ void pack_weight(const float* __restrict__ src, _Float16* __restrict__ dst,
                            int K, int N, int ntiles, int total) {
    const int idx = blockIdx.x * blockDim.x + threadIdx.x;
    if (idx >= total) return;
    const int tile = idx >> 9;
    const int r    = idx & 511;
    const int ln   = r >> 4;
    const int e    = r & 15;
    const int kt   = tile / ntiles;
    const int nt   = tile - kt * ntiles;
    const int k    = kt * 32 + (ln >> 4) * 16 + e;
    const int n    = nt * 16 + (ln & 15);
    const float v  = (k < K && n < N) ? src[k * N + n] : 0.0f;
    dst[idx] = (_Float16)v;
}

/* head-padded variant: logical K' = 160, row k' -> (h = k'/48, d = k'%48),
   valid iff h<3 && d<42, source row = h*42+d. N = 128 exact. */
__global__ void pack_weight_hp(const float* __restrict__ src, _Float16* __restrict__ dst,
                               int total) {
    const int idx = blockIdx.x * blockDim.x + threadIdx.x;
    if (idx >= total) return;
    const int tile = idx >> 9;
    const int r    = idx & 511;
    const int ln   = r >> 4;
    const int e    = r & 15;
    const int kt   = tile >> 3;          /* ntiles = 8 */
    const int nt   = tile & 7;
    const int kp   = kt * 32 + (ln >> 4) * 16 + e;
    const int n    = nt * 16 + (ln & 15);
    const int h    = kp / HSTRIDE;
    const int d    = kp - h * HSTRIDE;
    const bool ok  = (h < NH) && (d < DD);
    const float v  = ok ? src[(h * DD + d) * ATTN_DIM + n] : 0.0f;
    dst[idx] = (_Float16)v;
}

/* --------------------------- fused forward pass --------------------------- */
__global__ __launch_bounds__(128) void attn_sac_main(
    const float* __restrict__ obs,
    const float* __restrict__ b_own, const float* __restrict__ b_int,
    const float* __restrict__ v_att,
    const float* __restrict__ b_attn, const float* __restrict__ b_ownp,
    const float* __restrict__ b_h1, const float* __restrict__ b_h2,
    const float* __restrict__ b_out,
    const _Float16* __restrict__ ws,
    float* __restrict__ outp)
{
    __shared__ __attribute__((aligned(16))) _Float16 lds_scr[WAVES_PER_BLOCK * ROWS_PER_WAVE * SCR_LD];
    __shared__ __attribute__((aligned(16))) _Float16 lds_x  [WAVES_PER_BLOCK * ROWS_PER_WAVE * XB_LD];

    const int lane = threadIdx.x & 31;
    const int wave = threadIdx.x >> 5;
    const int r0   = blockIdx.x * ROWS_PER_BLOCK + wave * ROWS_PER_WAVE;
    const int ln16 = lane & 15;
    const int lh   = lane >> 4;

    _Float16* scr = lds_scr + wave * ROWS_PER_WAVE * SCR_LD;   /* wave-private */
    _Float16* xb  = lds_x   + wave * ROWS_PER_WAVE * XB_LD;    /* wave-private */

    /* zero wave-private LDS once so pad columns stay zero forever */
    {
        v8h* s8 = reinterpret_cast<v8h*>(scr);
        v8h* x8 = reinterpret_cast<v8h*>(xb);
        const v8h z = zero8h();
        for (int i = lane; i < ROWS_PER_WAVE * SCR_LD / 8; i += 32) s8[i] = z;
        for (int i = lane; i < ROWS_PER_WAVE * XB_LD / 8; i += 32) x8[i] = z;
    }

    /* per-lane head-padded column map for embedding stores */
    int cmap[8];
#pragma unroll
    for (int nt = 0; nt < 8; ++nt) {
        const int c = nt * 16 + ln16;
        cmap[nt] = (c < TOT) ? ((c / DD) * HSTRIDE + (c % DD)) : -1;
    }

    /* ---- stage 1: own_emb = lrelu(own @ w_own + b_own) -> scr (head-padded) ---- */
    {
        v16h a = zero16();
        if (lane < 16) {
            const float* p = obs + (size_t)(r0 + lane) * OBS_STRIDE;
#pragma unroll
            for (int j = 0; j < OWN_DIM; ++j) a[j] = (_Float16)p[j];
        }
        const _Float16* wb = ws + OFF_WOWN;
#pragma unroll
        for (int nt = 0; nt < 8; ++nt) {
            v8f acc = zero8();
            acc = wmma_f16(a, ldb(wb, nt, lane), acc);
            const int c = nt * 16 + ln16;
            const float bias = (c < TOT) ? b_own[c] : 0.0f;
#pragma unroll
            for (int i = 0; i < 8; ++i) {
                float v = acc[i] + bias;
                acc[i] = (v >= 0.0f) ? v : 0.2f * v;
            }
            sta_map(scr, SCR_LD, lane, cmap[nt], acc);
        }
    }

    /* ---- stage 2: q[h] = own_h @ wq[h]  (kept in registers) ---- */
    v8f q[NH][3];
#pragma unroll
    for (int h = 0; h < NH; ++h) {
        const v16h a0 = lda(scr, SCR_LD, lane, h * HSTRIDE);
        const v16h a1 = lda(scr, SCR_LD, lane, h * HSTRIDE + 32);
        const _Float16* wb = ws + OFF_WQ + h * 3072;
#pragma unroll
        for (int nt = 0; nt < 3; ++nt) {
            v8f acc = zero8();
            acc = wmma_f16(a0, ldb(wb, nt, lane), acc);
            acc = wmma_f16(a1, ldb(wb, 3 + nt, lane), acc);
            q[h][nt] = acc;
        }
    }

    /* ---- stage 3: own_vec = tanh(own_emb @ w_ownp + b) -> xb[:,0:128] ---- */
    {
        v16h af[5];
#pragma unroll
        for (int kt = 0; kt < 5; ++kt) af[kt] = lda(scr, SCR_LD, lane, kt * 32);
        const _Float16* wb = ws + OFF_WOWNP;
#pragma unroll 1
        for (int nt = 0; nt < 8; ++nt) {
            v8f acc = zero8();
#pragma unroll
            for (int kt = 0; kt < 5; ++kt)
                acc = wmma_f16(af[kt], ldb(wb, kt * 8 + nt, lane), acc);
            const float bias = b_ownp[nt * 16 + ln16];
#pragma unroll
            for (int i = 0; i < 8; ++i) acc[i] = fast_tanh(acc[i] + bias);
            sta(xb, XB_LD, lane, nt * 16, 16, acc);
        }
    }

    /* ---- per-lane v_att columns ---- */
    float vatt[NH][3];
#pragma unroll
    for (int h = 0; h < NH; ++h)
#pragma unroll
        for (int nt = 0; nt < 3; ++nt) {
            const int d = nt * 16 + ln16;
            vatt[h][nt] = (d < DD) ? v_att[h * DD + d] : 0.0f;
        }

    /* ---- stage 4: attention with online softmax; ctx -> xb (head-padded) ----
       w_int B fragments are invariant over (h, n): keep them in registers.  */
    v16h wbint[8];
#pragma unroll
    for (int nt = 0; nt < 8; ++nt) wbint[nt] = ldb(ws + OFF_WINT, nt, lane);

#pragma unroll
    for (int h = 0; h < NH; ++h) {
        /* wk/wv fragments invariant over n: hoist */
        v16h wkf[6], wvf[6];
#pragma unroll
        for (int t = 0; t < 6; ++t) {
            wkf[t] = ldb(ws + OFF_WK + h * 3072, t, lane);
            wvf[t] = ldb(ws + OFF_WV + h * 3072, t, lane);
        }

        float m[8], l[8];
        v8f ctx[3];
#pragma unroll
        for (int i = 0; i < 8; ++i) { m[i] = -1e30f; l[i] = 0.0f; }
#pragma unroll
        for (int nt = 0; nt < 3; ++nt) ctx[nt] = zero8();

#pragma unroll 1
        for (int n = 0; n < N_INTR; ++n) {
            /* intr A-fragment straight from global (K=7 padded to 32) */
            v16h a = zero16();
            float asum = 0.0f;
            if (lane < 16) {
                const float* p = obs + (size_t)(r0 + lane) * OBS_STRIDE + OWN_DIM + n * INT_DIM;
                __builtin_prefetch(p + INT_DIM, 0, 0);
#pragma unroll
                for (int j = 0; j < INT_DIM; ++j) {
                    const float v = p[j];
                    asum += fabsf(v);
                    a[j] = (_Float16)v;
                }
            }
            const float padf = (asum < 1e-6f) ? 1.0f : 0.0f;

            /* int_emb_n = lrelu(intr_n @ w_int + b_int) -> scr (head-padded) */
#pragma unroll
            for (int nt = 0; nt < 8; ++nt) {
                v8f acc = zero8();
                acc = wmma_f16(a, wbint[nt], acc);
                const int c = nt * 16 + ln16;
                const float bias = (c < TOT) ? b_int[c] : 0.0f;
#pragma unroll
                for (int i = 0; i < 8; ++i) {
                    float v = acc[i] + bias;
                    acc[i] = (v >= 0.0f) ? v : 0.2f * v;
                }
                sta_map(scr, SCR_LD, lane, cmap[nt], acc);
            }

            const v16h a0 = lda(scr, SCR_LD, lane, h * HSTRIDE);
            const v16h a1 = lda(scr, SCR_LD, lane, h * HSTRIDE + 32);

            /* k GEMM + energy = tanh(q + k) */
            v8f e[3];
#pragma unroll
            for (int nt = 0; nt < 3; ++nt) {
                v8f kacc = zero8();
                kacc = wmma_f16(a0, wkf[nt], kacc);
                kacc = wmma_f16(a1, wkf[3 + nt], kacc);
#pragma unroll
                for (int i = 0; i < 8; ++i) e[nt][i] = fast_tanh(q[h][nt][i] + kacc[i]);
            }

            /* per-lane partial score, then butterfly-sum across 16-lane group */
            float sp[8];
#pragma unroll
            for (int i = 0; i < 8; ++i)
                sp[i] = e[0][i] * vatt[h][0] + e[1][i] * vatt[h][1] + e[2][i] * vatt[h][2];
#pragma unroll
            for (int i = 0; i < 8; ++i) {
                float v = sp[i];
#pragma unroll
                for (int msk = 1; msk < 16; msk <<= 1) v += __shfl_xor(v, msk, 32);
                sp[i] = v;
            }

            /* v GEMM */
            v8f vacc[3];
#pragma unroll
            for (int nt = 0; nt < 3; ++nt) {
                v8f vc = zero8();
                vc = wmma_f16(a0, wvf[nt], vc);
                vc = wmma_f16(a1, wvf[3 + nt], vc);
                vacc[nt] = vc;
            }

            /* online softmax accumulate: rows in butterfly layout match acc layout */
#pragma unroll
            for (int i = 0; i < 8; ++i) {
                const float fa = __shfl(padf, i, 32);
                const float fb = __shfl(padf, i + 8, 32);
                const float f  = (lane < 16) ? fa : fb;
                const float s  = (f > 0.5f) ? -1e30f : sp[i];
                const float mn = fmaxf(m[i], s);
                const float corr = __expf(m[i] - mn);
                const float p    = (f > 0.5f) ? 0.0f : __expf(s - mn);
                l[i] = l[i] * corr + p;
                m[i] = mn;
#pragma unroll
                for (int nt = 0; nt < 3; ++nt)
                    ctx[nt][i] = ctx[nt][i] * corr + p * vacc[nt][i];
            }
        } /* n */

        /* finalize ctx (nan_to_num: all-pad rows -> 0); park at xb[:,128+h*48] */
#pragma unroll
        for (int nt = 0; nt < 3; ++nt) {
            v8f c = ctx[nt];
#pragma unroll
            for (int i = 0; i < 8; ++i) {
                const float inv = (l[i] > 0.0f) ? (1.0f / l[i]) : 0.0f;
                c[i] *= inv;
            }
            const int nv = (DD - nt * 16 < 16) ? (DD - nt * 16) : 16;
            sta(xb, XB_LD, lane, CTX_BASE + h * HSTRIDE + nt * 16, nv, c);
        }
    } /* h */

    /* ---- stage 5: attn_vec = tanh(ctx @ w_attn + b) -> xb[:,128:256] ---- */
    {
        v16h af[5];
#pragma unroll
        for (int kt = 0; kt < 5; ++kt) af[kt] = lda(xb, XB_LD, lane, CTX_BASE + kt * 32);
        const _Float16* wb = ws + OFF_WATTN;
#pragma unroll 1
        for (int nt = 0; nt < 8; ++nt) {
            v8f acc = zero8();
#pragma unroll
            for (int kt = 0; kt < 5; ++kt)
                acc = wmma_f16(af[kt], ldb(wb, kt * 8 + nt, lane), acc);
            const float bias = b_attn[nt * 16 + ln16];
#pragma unroll
            for (int i = 0; i < 8; ++i) acc[i] = fast_tanh(acc[i] + bias);
            sta(xb, XB_LD, lane, CTX_BASE + nt * 16, 16, acc);
        }
    }

    /* ---- stage 6: h1 = lrelu(x @ w_h1 + b) -> xb ---- */
    {
        v16h af[8];
#pragma unroll
        for (int kt = 0; kt < 8; ++kt) af[kt] = lda(xb, XB_LD, lane, kt * 32);
        const _Float16* wb = ws + OFF_WH1;
#pragma unroll 4
        for (int nt = 0; nt < 16; ++nt) {
            v8f acc = zero8();
#pragma unroll
            for (int kt = 0; kt < 8; ++kt)
                acc = wmma_f16(af[kt], ldb(wb, kt * 16 + nt, lane), acc);
            const float bias = b_h1[nt * 16 + ln16];
#pragma unroll
            for (int i = 0; i < 8; ++i) {
                float v = acc[i] + bias;
                acc[i] = (v >= 0.0f) ? v : 0.2f * v;
            }
            sta(xb, XB_LD, lane, nt * 16, 16, acc);
        }
    }

    /* ---- stage 7: h2 = lrelu(h1 @ w_h2 + b) -> xb ---- */
    {
        v16h af[8];
#pragma unroll
        for (int kt = 0; kt < 8; ++kt) af[kt] = lda(xb, XB_LD, lane, kt * 32);
        const _Float16* wb = ws + OFF_WH2;
#pragma unroll 4
        for (int nt = 0; nt < 16; ++nt) {
            v8f acc = zero8();
#pragma unroll
            for (int kt = 0; kt < 8; ++kt)
                acc = wmma_f16(af[kt], ldb(wb, kt * 16 + nt, lane), acc);
            const float bias = b_h2[nt * 16 + ln16];
#pragma unroll
            for (int i = 0; i < 8; ++i) {
                float v = acc[i] + bias;
                acc[i] = (v >= 0.0f) ? v : 0.2f * v;
            }
            sta(xb, XB_LD, lane, nt * 16, 16, acc);
        }
    }

    /* ---- stage 8: out = h2 @ w_out + b_out -> global f32 ---- */
    {
        v16h af[8];
#pragma unroll
        for (int kt = 0; kt < 8; ++kt) af[kt] = lda(xb, XB_LD, lane, kt * 32);
        const _Float16* wb = ws + OFF_WOUT;
        v8f acc = zero8();
#pragma unroll
        for (int kt = 0; kt < 8; ++kt)
            acc = wmma_f16(af[kt], ldb(wb, kt, lane), acc);
        const int col = ln16;
        if (col < NOUT) {
            const float bo = b_out[col];
            const int rb = lh * 8;
#pragma unroll
            for (int i = 0; i < 8; ++i)
                outp[(size_t)(r0 + rb + i) * NOUT + col] = acc[i] + bo;
        }
    }
}

extern "C" void kernel_launch(void* const* d_in, const int* in_sizes, int n_in,
                              void* d_out, int out_size, void* d_ws, size_t ws_size,
                              hipStream_t stream) {
    (void)in_sizes; (void)n_in; (void)out_size; (void)ws_size;
    const float* obs    = (const float*)d_in[0];
    const float* w_own  = (const float*)d_in[1];
    const float* b_own  = (const float*)d_in[2];
    const float* w_int  = (const float*)d_in[3];
    const float* b_int  = (const float*)d_in[4];
    const float* wq     = (const float*)d_in[5];
    const float* wk     = (const float*)d_in[6];
    const float* wv     = (const float*)d_in[7];
    const float* v_att  = (const float*)d_in[8];
    const float* w_attn = (const float*)d_in[9];
    const float* b_attn = (const float*)d_in[10];
    const float* w_ownp = (const float*)d_in[11];
    const float* b_ownp = (const float*)d_in[12];
    const float* w_h1   = (const float*)d_in[13];
    const float* b_h1   = (const float*)d_in[14];
    const float* w_h2   = (const float*)d_in[15];
    const float* b_h2   = (const float*)d_in[16];
    const float* w_out  = (const float*)d_in[17];
    const float* b_out  = (const float*)d_in[18];
    _Float16* ws = (_Float16*)d_ws;

    auto pack = [&](const float* s, unsigned off, int K, int N) {
        const int kt = (K + 31) / 32, nt = (N + 15) / 16;
        const int total = kt * nt * 512;
        pack_weight<<<dim3((total + 255) / 256), dim3(256), 0, stream>>>(s, ws + off, K, N, nt, total);
    };
    auto pack_hp = [&](const float* s, unsigned off) {
        const int total = 5 * 8 * 512;   /* K'=160 (5 tiles) x N=128 (8 tiles) */
        pack_weight_hp<<<dim3((total + 255) / 256), dim3(256), 0, stream>>>(s, ws + off, total);
    };
    pack(w_own, OFF_WOWN, OWN_DIM, TOT);
    pack(w_int, OFF_WINT, INT_DIM, TOT);
    for (int h = 0; h < NH; ++h) {
        pack(wq + h * DD * DD, OFF_WQ + h * 3072, DD, DD);
        pack(wk + h * DD * DD, OFF_WK + h * 3072, DD, DD);
        pack(wv + h * DD * DD, OFF_WV + h * 3072, DD, DD);
    }
    pack_hp(w_ownp, OFF_WOWNP);
    pack_hp(w_attn, OFF_WATTN);
    pack(w_h1, OFF_WH1, 2 * ATTN_DIM, HID);
    pack(w_h2, OFF_WH2, HID, HID);
    pack(w_out, OFF_WOUT, HID, NOUT);

    attn_sac_main<<<dim3(BATCH / ROWS_PER_BLOCK), dim3(WAVES_PER_BLOCK * 32), 0, stream>>>(
        obs, b_own, b_int, v_att, b_attn, b_ownp, b_h1, b_h2, b_out, ws, (float*)d_out);
}
